// SelfAttention_83270825935110
// MI455X (gfx1250) — matrix operand order repaired and checked by reading, not compiled, not run on hardware
//
#include <hip/hip_runtime.h>
#include <hip/hip_bf16.h>

#define BB 4
#define CC 256
#define NN 4096
#define DD 32

typedef __attribute__((ext_vector_type(16))) __bf16          v16bf;
typedef __attribute__((ext_vector_type(16))) unsigned short  v16u;
typedef __attribute__((ext_vector_type(8)))  unsigned short  v8u;
typedef __attribute__((ext_vector_type(8)))  unsigned int    v8ui;
typedef __attribute__((ext_vector_type(8)))  float           v8f;

__device__ __forceinline__ unsigned short f2bf(float f) {
  unsigned int u = __builtin_bit_cast(unsigned int, f);
  u += 0x7FFFu + ((u >> 16) & 1u);           // round-to-nearest-even
  return (unsigned short)(u >> 16);
}
__device__ __forceinline__ unsigned int pack2bf(float lo, float hi) {
  return (unsigned int)f2bf(lo) | ((unsigned int)f2bf(hi) << 16);
}

// Generic LDS pointer = {SHARED_BASE, offset}; low 32 bits are the LDS byte addr.
__device__ __forceinline__ unsigned int lds_addr32(const void* p) {
  return (unsigned int)(unsigned long long)p;
}
// Async DMA global->LDS (ASYNCcnt-tracked), gfx1250.
__device__ __forceinline__ void async_ld_b128(unsigned int lds, const void* g) {
  asm volatile("global_load_async_to_lds_b128 %0, %1, off"
               :: "v"(lds), "v"((unsigned long long)(size_t)g) : "memory");
}
__device__ __forceinline__ void async_ld_b64(unsigned int lds, const void* g) {
  asm volatile("global_load_async_to_lds_b64 %0, %1, off"
               :: "v"(lds), "v"((unsigned long long)(size_t)g) : "memory");
}

// ---------------------------------------------------------------------------
// Projection kernel: q,k,v 1x1 convs; q and k stored transposed [B][N][D].
// grid.x = B*320, block = 256.
// ---------------------------------------------------------------------------
__global__ void proj_kernel(const float* __restrict__ x,
                            const float* __restrict__ wq, const float* __restrict__ bq,
                            const float* __restrict__ wk, const float* __restrict__ bk,
                            const float* __restrict__ wv, const float* __restrict__ bv,
                            unsigned short* __restrict__ qT,   // [B][N][D] bf16
                            unsigned short* __restrict__ kT,   // [B][N][D] bf16
                            unsigned short* __restrict__ vbf)  // [B][C][N] bf16
{
  __shared__ float wrow[CC];
  const int b = blockIdx.x / 320;
  const int r = blockIdx.x % 320;

  const float* w;
  float bias;
  if (r < DD)            { w = wq + (size_t)r * CC;          bias = bq[r]; }
  else if (r < 2 * DD)   { w = wk + (size_t)(r - DD) * CC;   bias = bk[r - DD]; }
  else                   { w = wv + (size_t)(r - 2*DD) * CC; bias = bv[r - 2*DD]; }

  wrow[threadIdx.x] = w[threadIdx.x];
  __syncthreads();

  const float* xb = x + (size_t)b * CC * NN;
  for (int nn = 0; nn < NN / 256; ++nn) {
    const int n = nn * 256 + threadIdx.x;
    float acc = bias;
#pragma unroll 8
    for (int c = 0; c < CC; ++c)
      acc = fmaf(wrow[c], xb[(size_t)c * NN + n], acc);
    const unsigned short bfv = f2bf(acc);
    if (r < DD)          qT [((size_t)b * NN + n) * DD + r]          = bfv;
    else if (r < 2 * DD) kT [((size_t)b * NN + n) * DD + (r - DD)]   = bfv;
    else                 vbf[((size_t)b * CC + (r - 2*DD)) * NN + n] = bfv;
  }
}

// ---------------------------------------------------------------------------
// Fused flash-attention kernel, transposed-score formulation:
//   S^T = K (j x d) * Q (d x i)   ;   O' = V (c x j) * P^T (j x i)
// Double-buffered K/V tiles streamed via global_load_async_to_lds (ASYNCcnt):
// chunk jc+1 is in flight while chunk jc feeds 18 v_wmma per wave.
// block = 256 threads = 8 waves; each wave: 16 query rows, 16x256 f32 acc.
// grid = (N/128, B).
// ---------------------------------------------------------------------------
__global__ void __launch_bounds__(256) attn_kernel(
    const float* __restrict__ x, const float* __restrict__ gammap,
    const unsigned short* __restrict__ qT,
    const unsigned short* __restrict__ kT,
    const unsigned short* __restrict__ vbf,
    float* __restrict__ out)
{
  __shared__ alignas(64) unsigned short kt[2][32][DD];     // K chunks [j][d]
  __shared__ alignas(64) unsigned short vt[2][CC][32];     // V chunks [c][j]

  const int tid   = threadIdx.x;
  const int wave  = tid >> 5;
  const int lane  = tid & 31;
  const int n     = lane & 15;     // query column / tile column
  const int h     = lane >> 4;     // half-wave selector
  const int b     = blockIdx.y;
  const int ibase = blockIdx.x * 128 + wave * 16;

  const unsigned short* kTb = kT  + (size_t)b * NN * DD;
  const unsigned short* vb  = vbf + (size_t)b * CC * NN;

  // Per-thread staging coordinates (5 async ops/chunk/thread, wave-uniform).
  const int kj = tid >> 3, kd = (tid & 7) * 4;             // K: 1 x b64
  const int vc = tid >> 2, vp = (tid & 3) * 8;             // V: 4 x b128 (c += 64)

  auto stage = [&](int chunk, int buf) {
    const int j0 = chunk * 32;
    async_ld_b64(lds_addr32(&kt[buf][kj][kd]),
                 kTb + (size_t)(j0 + kj) * DD + kd);
#pragma unroll
    for (int it = 0; it < 4; ++it) {
      const int c = vc + 64 * it;
      async_ld_b128(lds_addr32(&vt[buf][c][vp]),
                    vb + (size_t)c * NN + j0 + vp);
    }
  };

  // ---- Q as WMMA B-operand (d x i), loaded once ----
  const v16u qu = *(const v16u*)(qT + ((size_t)b * NN + (ibase + n)) * DD + 16 * h);
  const v16bf qb = __builtin_bit_cast(v16bf, qu);

  const v8f zero = {0.f, 0.f, 0.f, 0.f, 0.f, 0.f, 0.f, 0.f};
  v8f O[16];
#pragma unroll
  for (int t = 0; t < 16; ++t) O[t] = zero;

  float mrun = -3.0e38f, lrun = 0.0f;

  const int NCH = NN / 32;
  stage(0, 0);
  stage(1, 1);

#pragma unroll 1
  for (int jc = 0; jc < NCH; ++jc) {
    const int cur = jc & 1;

    // oldest in-flight chunk (jc) has landed; chunk jc+1 may still be in flight
    if (jc + 1 < NCH) asm volatile("s_wait_asynccnt 0x5" ::: "memory");
    else              asm volatile("s_wait_asynccnt 0x0" ::: "memory");
    __syncthreads();   // all waves' chunk-jc DMA visible

    // ---- S^T tiles: A = K rows (16 j x 32 d), B = Q ----
    v16u a0, a1;
    {
      const v8u lo0 = *(const v8u*)&kt[cur][n][16 * h];
      const v8u hi0 = *(const v8u*)&kt[cur][n][16 * h + 8];
      const v8u lo1 = *(const v8u*)&kt[cur][16 + n][16 * h];
      const v8u hi1 = *(const v8u*)&kt[cur][16 + n][16 * h + 8];
      a0 = __builtin_shufflevector(lo0, hi0, 0,1,2,3,4,5,6,7,8,9,10,11,12,13,14,15);
      a1 = __builtin_shufflevector(lo1, hi1, 0,1,2,3,4,5,6,7,8,9,10,11,12,13,14,15);
    }
    v8f S0 = __builtin_amdgcn_wmma_f32_16x16x32_bf16(
        false, __builtin_bit_cast(v16bf, a0), false, qb, (short)0, zero, false, false);
    v8f S1 = __builtin_amdgcn_wmma_f32_16x16x32_bf16(
        false, __builtin_bit_cast(v16bf, a1), false, qb, (short)0, zero, false, false);

    // ---- online softmax over the 32 keys of this chunk ----
    float mx = S0[0];
#pragma unroll
    for (int r = 1; r < 8; ++r) mx = fmaxf(mx, S0[r]);
#pragma unroll
    for (int r = 0; r < 8; ++r) mx = fmaxf(mx, S1[r]);
    mx = fmaxf(mx, __shfl_xor(mx, 16, 32));
    const float mnew = fmaxf(mrun, mx);
    const float sc   = __expf(mrun - mnew);
    mrun = mnew;

    float P0[8], P1[8], rs = 0.0f;
#pragma unroll
    for (int r = 0; r < 8; ++r) {
      P0[r] = __expf(S0[r] - mnew);
      P1[r] = __expf(S1[r] - mnew);
      rs += P0[r] + P1[r];
    }
    rs += __shfl_xor(rs, 16, 32);
    lrun = lrun * sc + rs;

#pragma unroll
    for (int t = 0; t < 16; ++t) {
#pragma unroll
      for (int r = 0; r < 8; ++r) O[t][r] *= sc;
    }

    // ---- build P^T B-operand in registers: 4 packed-bf16 xor-16 shuffles ----
    unsigned int p0p[4], p1p[4], recv[4];
#pragma unroll
    for (int p = 0; p < 4; ++p) {
      p0p[p] = pack2bf(P0[2 * p], P0[2 * p + 1]);
      p1p[p] = pack2bf(P1[2 * p], P1[2 * p + 1]);
      const unsigned int send = h ? p0p[p] : p1p[p];
      recv[p] = (unsigned int)__shfl_xor((int)send, 16, 32);
    }
    v8ui bbu;
#pragma unroll
    for (int p = 0; p < 4; ++p) {
      bbu[p]     = h ? recv[p] : p0p[p];   // rows K = 16h + (0..7)
      bbu[p + 4] = h ? p1p[p]  : recv[p];  // rows K = 16h + (8..15)
    }
    const v16bf pb = __builtin_bit_cast(v16bf, bbu);

    // ---- O' += V x P^T : 16 channel tiles ----
#pragma unroll
    for (int t = 0; t < 16; ++t) {
      const v8u lo = *(const v8u*)&vt[cur][16 * t + n][16 * h];
      const v8u hi = *(const v8u*)&vt[cur][16 * t + n][16 * h + 8];
      const v16u av = __builtin_shufflevector(lo, hi, 0,1,2,3,4,5,6,7,8,9,10,11,12,13,14,15);
      O[t] = __builtin_amdgcn_wmma_f32_16x16x32_bf16(
          false, __builtin_bit_cast(v16bf, av), false, pb, (short)0, O[t], false, false);
    }

    __syncthreads();   // all waves done reading buf[cur]
    if (jc + 2 < NCH) stage(jc + 2, cur);   // refill the freed buffer
  }

  // ---- epilogue: normalize, gamma*out + x (queries in lanes -> coalesced) ----
  const float gamma = gammap[0];
  const float inv   = 1.0f / lrun;

  const float* xb2 = x   + (size_t)b * CC * NN;
  float*       ob  = out + (size_t)b * CC * NN;
#pragma unroll
  for (int t = 0; t < 16; ++t) {
#pragma unroll
    for (int r = 0; r < 8; ++r) {
      const int c = 16 * t + r + 8 * h;
      const size_t idx = (size_t)c * NN + (ibase + n);
      ob[idx] = gamma * (O[t][r] * inv) + xb2[idx];
    }
  }
}

// ---------------------------------------------------------------------------
extern "C" void kernel_launch(void* const* d_in, const int* in_sizes, int n_in,
                              void* d_out, int out_size, void* d_ws, size_t ws_size,
                              hipStream_t stream) {
  (void)in_sizes; (void)n_in; (void)out_size; (void)ws_size;
  const float* x     = (const float*)d_in[0];
  const float* wq    = (const float*)d_in[1];
  const float* bq    = (const float*)d_in[2];
  const float* wk    = (const float*)d_in[3];
  const float* bk    = (const float*)d_in[4];
  const float* wv    = (const float*)d_in[5];
  const float* bv    = (const float*)d_in[6];
  const float* gamma = (const float*)d_in[7];
  float* out = (float*)d_out;

  // workspace: bf16 qT [B][N][D], kT [B][N][D], v [B][C][N]  (~10.5 MB)
  unsigned short* qT  = (unsigned short*)d_ws;
  unsigned short* kT  = qT + (size_t)BB * NN * DD;
  unsigned short* vbf = kT + (size_t)BB * NN * DD;

  proj_kernel<<<dim3(BB * 320), dim3(256), 0, stream>>>(
      x, wq, bq, wk, bk, wv, bv, qT, kT, vbf);

  attn_kernel<<<dim3(NN / 128, BB), dim3(256), 0, stream>>>(
      x, gamma, qT, kT, vbf, out);
}
